// QuantizableCrossAttention_63642825392134
// MI455X (gfx1250) — compile-verified, matrix operand order checked by reading, not run on hardware
//
#include <hip/hip_runtime.h>
#include <hip/hip_bf16.h>

// ---------------------------------------------------------------------------
// CDNA5 (gfx1250) bf16 WMMA cross-attention pipeline with TDM K-staging.
//   B=8, N=4096, M=512, Q_DIM=1024, KV_DIM=768, HEADS=8, DIM_HEAD=128, INNER=1024
// ---------------------------------------------------------------------------

typedef __attribute__((ext_vector_type(16))) __bf16    v16bf;
typedef __attribute__((ext_vector_type(8)))  float     v8f;
typedef __attribute__((ext_vector_type(4)))  unsigned  v4u;
typedef __attribute__((ext_vector_type(8)))  int       v8i;
typedef __attribute__((ext_vector_type(4)))  int       v4i;

#if __has_builtin(__builtin_amdgcn_tensor_load_to_lds) && \
    __has_builtin(__builtin_amdgcn_s_wait_tensorcnt)
#define USE_TDM 1
#else
#define USE_TDM 0
#endif

__device__ __forceinline__ v8f wmma_bf16(v16bf a, v16bf b, v8f c) {
  // 8 args: (neg_a, A, neg_b, B, c_mod, C, reuse_a, reuse_b)
  return __builtin_amdgcn_wmma_f32_16x16x32_bf16(false, a, false, b, (short)0, c,
                                                 false, false);
}

// Load a 16x32 bf16 A/B fragment from a row-major [16 x ld] bf16 buffer.
// ISA 7.12.2: lane L(0-15)=row L, halves hold K={0..7,16..23}; lanes 16-31
// hold K={8..15,24..31}. B fragments are identical with "row"=output column,
// provided the buffer is [col][k] contiguous in k.
__device__ __forceinline__ v16bf load_frag16(const __bf16* __restrict__ base, int ld) {
  const int lane  = threadIdx.x & 31;
  const int row   = lane & 15;
  const int khalf = (lane >> 4) << 3;            // 0 or 8
  const __bf16* p = base + (size_t)row * ld + khalf;
  v16bf f;
#pragma unroll
  for (int e = 0; e < 16; ++e) {
    const int pr = e >> 1;
    const int k  = ((pr & 3) << 1) + ((pr >> 2) << 4) + (e & 1);
    f[e] = p[k];
  }
  return f;
}

#if USE_TDM
// Issue one TDM 2D tile load: 64 rows x 128 bf16 elements, row stride = 1024
// elements (one head slice out of K [kv][INNER]).  Descriptor packing per
// CDNA5 ISA 8.3/8.4 (D# group0: count/lds/global/type; group1: dims+strides).
// This toolchain's builtin takes 6 args (g0, g1, g2, g3, g4-extra, cpol).
__device__ __forceinline__ void tdm_load_tile_64x128(const void* gsrc,
                                                     unsigned lds_off) {
  const unsigned long long ga = (unsigned long long)(__SIZE_TYPE__)gsrc;
  v4u g0;
  g0[0] = 1u;                                       // count=1, user mode
  g0[1] = lds_off;                                  // lds_addr (bytes)
  g0[2] = (unsigned)(ga & 0xFFFFFFFFu);             // global_addr[31:0]
  g0[3] = (unsigned)((ga >> 32) & 0x01FFFFFFu)      // global_addr[56:32]
          | (2u << 30);                             // type=2 ("image")
  v8i g1;
  g1[0] = (int)(1u << 16);        // wg_mask=0 | data_size=1 (2 bytes)
  g1[1] = (int)(128u << 16);      // tensor_dim0[15:0]=128 (d)
  g1[2] = (int)(64u << 16);       // tensor_dim0 hi=0 | tensor_dim1 lo=64 (kv)
  g1[3] = (int)(128u << 16);      // tensor_dim1 hi=0 | tile_dim0=128
  g1[4] = 64;                     // tile_dim1=64 | tile_dim2=0
  g1[5] = 1024;                   // tensor_dim0_stride = 1024 elements
  g1[6] = 0;                      // stride hi / tensor_dim1_stride lo
  g1[7] = 0;
  const v4i z4 = {0, 0, 0, 0};                      // groups 2/3 unused (2D)
  const v8i z8 = {0, 0, 0, 0, 0, 0, 0, 0};          // extra group (zeroed)
  __builtin_amdgcn_tensor_load_to_lds(g0, g1, z4, z4, z8, 0);
}
#endif

// ---------------------------------------------------------------------------
// Generic tiled GEMM:  C[R x Nc] = A[R x K] * W[K x Nc]  (+ bias)
//   Block tile 128x128, K-step 32, 256 threads = 8 waves (4x2 wave grid),
//   each wave owns a 32x64 sub-tile = 2x4 WMMA accumulators.
// ---------------------------------------------------------------------------
template <bool A_IS_BF16, bool OUT_IS_F32>
__global__ __launch_bounds__(256) void gemm128_wmma(
    const void* __restrict__ Aptr, const float* __restrict__ W,
    const float* __restrict__ bias, void* __restrict__ Cptr,
    int R, int K, int Nc) {
  constexpr int BM = 128, BN = 128, BK = 32, LDT = BK + 2;
  __shared__ __bf16 As[BM * LDT];   // [row][k]
  __shared__ __bf16 Bs[BN * LDT];   // [col][k]  (W staged transposed)

  const int tid  = threadIdx.x;
  const int lane = tid & 31;
  const int wave = tid >> 5;
  const int wm   = wave >> 1;
  const int wn   = wave & 1;
  const int row0 = blockIdx.y * BM;
  const int col0 = blockIdx.x * BN;

  const v8f vzero = {0.f, 0.f, 0.f, 0.f, 0.f, 0.f, 0.f, 0.f};
  v8f acc[2][4];
#pragma unroll
  for (int i = 0; i < 2; ++i)
#pragma unroll
    for (int j = 0; j < 4; ++j) acc[i][j] = vzero;

  for (int k0 = 0; k0 < K; k0 += BK) {
    __syncthreads();
#pragma unroll
    for (int it = 0; it < 16; ++it) {
      const int idx = tid + (it << 8);
      const int r   = idx >> 5;
      const int kk  = idx & 31;
      float a;
      if (A_IS_BF16)
        a = (float)((const __bf16*)Aptr)[(size_t)(row0 + r) * K + (k0 + kk)];
      else
        a = ((const float*)Aptr)[(size_t)(row0 + r) * K + (k0 + kk)];
      As[r * LDT + kk] = (__bf16)a;
    }
#pragma unroll
    for (int it = 0; it < 16; ++it) {
      const int idx = tid + (it << 8);
      const int kk  = idx >> 7;
      const int n   = idx & 127;
      Bs[n * LDT + kk] = (__bf16)W[(size_t)(k0 + kk) * Nc + (col0 + n)];
    }
    if (k0 + BK < K) {   // L2 prefetch of next K tile (global_prefetch_b8)
      if (A_IS_BF16)
        __builtin_prefetch((const __bf16*)Aptr +
                               (size_t)(row0 + (tid >> 1)) * K + (k0 + BK), 0, 1);
      else
        __builtin_prefetch((const float*)Aptr +
                               (size_t)(row0 + (tid >> 1)) * K + (k0 + BK), 0, 1);
      __builtin_prefetch(W + (size_t)(k0 + BK + (tid & 31)) * Nc + col0, 0, 1);
    }
    __syncthreads();

    v16bf afrag[2], bfrag[4];
#pragma unroll
    for (int i = 0; i < 2; ++i)
      afrag[i] = load_frag16(&As[(wm * 32 + i * 16) * LDT], LDT);
#pragma unroll
    for (int j = 0; j < 4; ++j)
      bfrag[j] = load_frag16(&Bs[(wn * 64 + j * 16) * LDT], LDT);
#pragma unroll
    for (int i = 0; i < 2; ++i)
#pragma unroll
      for (int j = 0; j < 4; ++j)
        acc[i][j] = wmma_bf16(afrag[i], bfrag[j], acc[i][j]);
  }

  const int lr = lane & 15, lh = lane >> 4;
#pragma unroll
  for (int i = 0; i < 2; ++i)
#pragma unroll
    for (int j = 0; j < 4; ++j)
#pragma unroll
      for (int r = 0; r < 8; ++r) {
        const int grow = row0 + wm * 32 + i * 16 + r + (lh << 3);
        const int gcol = col0 + wn * 64 + j * 16 + lr;
        const float v  = acc[i][j][r];
        if (OUT_IS_F32)
          ((float*)Cptr)[(size_t)grow * Nc + gcol] = v + bias[gcol];
        else
          ((__bf16*)Cptr)[(size_t)grow * Nc + gcol] = (__bf16)v;
      }
}

// ---------------------------------------------------------------------------
// Fused flash-attention, M=512 kv in 8 chunks of 64, DIM_HEAD=128.
// Grid: (N/128, HEADS, B); 256 threads = 8 waves; each wave owns 16 Q rows.
//   - Q fragments live in registers for the whole kernel.
//   - K chunk (64x128 bf16) staged in LDS by the Tensor Data Mover,
//     double-buffered: wave 0 issues tensor_load_to_lds for chunk c+1 while
//     all waves compute chunk c; s_wait_tensorcnt + barrier synchronize.
//   - V chunk staged transposed ([d][kv]) so PV B-fragments read
//     kv-contiguous pairs.
//   - Online softmax via wave32 shfl_xor reductions over the C layout.
// LDS: K 2x16KB + Vt 16KB + P strips 8KB = 56KB.
// ---------------------------------------------------------------------------
__global__ __launch_bounds__(256) void attn_fused(
    const __bf16* __restrict__ Q, const __bf16* __restrict__ Kb,
    const __bf16* __restrict__ Vb, __bf16* __restrict__ O) {
  constexpr int INNER = 1024, D = 128, CH = 64, LV = CH, NCH = 512 / CH;
  __shared__ __bf16 Ks[2][CH * D];       // 2 x 16 KB, row-major [kv][d]
  __shared__ __bf16 Vt[D * LV];          // 16 KB, transposed [d][kv]
  __shared__ __bf16 Ps[8 * 16 * 32];     // 8 KB, per-wave 16x32 P slice

  const int tid = threadIdx.x, lane = tid & 31, wave = tid >> 5;
  const int lr = lane & 15, lh = lane >> 4;
  const int nt = blockIdx.x, h = blockIdx.y, b = blockIdx.z;
  const float SCALE = 0.08838834764831845f;  // 128^-0.5

  const size_t qrow0 = (size_t)b * 4096 + (size_t)nt * 128 + wave * 16;
  const __bf16* Qp = Q + qrow0 * INNER + h * D;
  const __bf16* Kp = Kb + (size_t)b * 512 * INNER + h * D;
  const __bf16* Vp = Vb + (size_t)b * 512 * INNER + h * D;

  v16bf qfrag[4];
#pragma unroll
  for (int kd = 0; kd < 4; ++kd) qfrag[kd] = load_frag16(Qp + kd * 32, INNER);

  const v8f vzero = {0.f, 0.f, 0.f, 0.f, 0.f, 0.f, 0.f, 0.f};
  v8f oacc[8];
#pragma unroll
  for (int j = 0; j < 8; ++j) oacc[j] = vzero;
  float mrun[8], lrun[8];
#pragma unroll
  for (int r = 0; r < 8; ++r) { mrun[r] = -3.0e38f; lrun[r] = 0.f; }

#if USE_TDM
  if (wave == 0)  // prologue: DMA chunk 0 into buffer 0
    tdm_load_tile_64x128(Kp, (unsigned)(__SIZE_TYPE__)&Ks[0][0]);
#endif

#pragma unroll
  for (int c = 0; c < NCH; ++c) {
    const int buf = c & 1;
    __syncthreads();  // prior chunk's LDS reads (Ks[buf ^ 1], Vt) are complete

#if USE_TDM
    if (wave == 0) {
      if (c + 1 < NCH)  // overlap: DMA next K chunk into the other buffer
        tdm_load_tile_64x128(Kp + (size_t)(c + 1) * CH * INNER,
                             (unsigned)(__SIZE_TYPE__)&Ks[buf ^ 1][0]);
    }
#else
    // Fallback: cooperative copy of K chunk (64x128) into LDS.
#pragma unroll
    for (int it = 0; it < 16; ++it) {
      const int idx = tid + (it << 8);  // 0..4095 dword pairs
      ((unsigned*)&Ks[buf][0])[idx] = *(
          (const unsigned*)(Kp + (size_t)(c * CH + (idx >> 6)) * INNER +
                            ((idx & 63) << 1)));
    }
#endif

    // Stage V chunk transposed: Vt[d][kv]; coalesced reads over d.
#pragma unroll
    for (int it = 0; it < 16; ++it) {
      const int idx = tid + (it << 8);  // 0..4095 bf16 pairs
      const int kv  = idx >> 6;
      const int dp  = (idx & 63) << 1;
      const __bf16* src = Vp + (size_t)(c * CH + kv) * INNER + dp;
      Vt[dp * LV + kv]       = src[0];
      Vt[(dp + 1) * LV + kv] = src[1];
    }

#if USE_TDM
    if (wave == 0) {
      if (c + 1 < NCH)
        __builtin_amdgcn_s_wait_tensorcnt(1);  // chunk c's DMA has landed
      else
        __builtin_amdgcn_s_wait_tensorcnt(0);
    }
#endif
    __syncthreads();

    // S = Q * K^T for this wave's 16 rows x 64 kv columns (4 col tiles).
    v8f s[4];
#pragma unroll
    for (int j = 0; j < 4; ++j) {
      v8f a = vzero;
#pragma unroll
      for (int kd = 0; kd < 4; ++kd) {
        v16bf kf = load_frag16(&Ks[buf][(j * 16) * D + kd * 32], D);
        a = wmma_bf16(qfrag[kd], kf, a);
      }
      s[j] = a;
    }

    // Online softmax. Lane groups (0-15 | 16-31) share rows 0-7 | 8-15.
    float pm[8];
#pragma unroll
    for (int r = 0; r < 8; ++r) pm[r] = -3.0e38f;
#pragma unroll
    for (int j = 0; j < 4; ++j)
#pragma unroll
      for (int r = 0; r < 8; ++r) {
        const float x = s[j][r] * SCALE;
        s[j][r] = x;
        pm[r] = fmaxf(pm[r], x);
      }
    for (int off = 1; off < 16; off <<= 1)
#pragma unroll
      for (int r = 0; r < 8; ++r)
        pm[r] = fmaxf(pm[r], __shfl_xor(pm[r], off, 32));

    float esc[8], psum[8];
#pragma unroll
    for (int r = 0; r < 8; ++r) {
      const float mn = fmaxf(mrun[r], pm[r]);
      esc[r]  = __expf(mrun[r] - mn);
      mrun[r] = mn;
      psum[r] = 0.f;
    }
#pragma unroll
    for (int j = 0; j < 4; ++j)
#pragma unroll
      for (int r = 0; r < 8; ++r) {
        const float p = __expf(s[j][r] - mrun[r]);
        s[j][r] = p;
        psum[r] += p;
      }
    for (int off = 1; off < 16; off <<= 1)
#pragma unroll
      for (int r = 0; r < 8; ++r) psum[r] += __shfl_xor(psum[r], off, 32);
#pragma unroll
    for (int r = 0; r < 8; ++r) lrun[r] = lrun[r] * esc[r] + psum[r];
#pragma unroll
    for (int j = 0; j < 8; ++j)
#pragma unroll
      for (int r = 0; r < 8; ++r) oacc[j][r] *= esc[r];

    // O += P * V. Per 32-wide k-step: relayout P slice (C->A layout) through
    // the per-wave LDS strip (wave-local, no barrier), then 8 d-tile WMMAs.
    __bf16* Pw = &Ps[wave * 16 * 32];
#pragma unroll
    for (int kd = 0; kd < 2; ++kd) {
#pragma unroll
      for (int t = 0; t < 2; ++t) {
        const int j = kd * 2 + t;
#pragma unroll
        for (int r = 0; r < 8; ++r)
          Pw[(r + (lh << 3)) * 32 + t * 16 + lr] = (__bf16)s[j][r];
      }
      v16bf pf = load_frag16(Pw, 32);
#pragma unroll
      for (int jd = 0; jd < 8; ++jd) {
        v16bf vf = load_frag16(&Vt[(jd * 16) * LV + kd * 32], LV);
        oacc[jd] = wmma_bf16(pf, vf, oacc[jd]);
      }
    }
  }

  // Normalize and write attn output (bf16) back to [B, N, INNER].
  __bf16* Op = O + qrow0 * INNER + h * D;
  float rinv[8];
#pragma unroll
  for (int r = 0; r < 8; ++r) rinv[r] = 1.f / lrun[r];
#pragma unroll
  for (int j = 0; j < 8; ++j)
#pragma unroll
    for (int r = 0; r < 8; ++r)
      Op[(size_t)(r + (lh << 3)) * INNER + j * 16 + lr] =
          (__bf16)(oacc[j][r] * rinv[r]);
}

// ---------------------------------------------------------------------------
// Host-side launch. Workspace (bf16 intermediates, ~151 MB):
//   qb: [8,4096,1024] 64MB | kb: [8,512,1024] 8MB | vb: 8MB | ab: 64MB
// ---------------------------------------------------------------------------
extern "C" void kernel_launch(void* const* d_in, const int* in_sizes, int n_in,
                              void* d_out, int out_size, void* d_ws,
                              size_t ws_size, hipStream_t stream) {
  const float* hidden = (const float*)d_in[0];  // [8,4096,1024]
  const float* enc    = (const float*)d_in[1];  // [8,512,768]
  const float* Wq     = (const float*)d_in[2];  // [1024,1024]
  const float* Wk     = (const float*)d_in[3];  // [768,1024]
  const float* Wv     = (const float*)d_in[4];  // [768,1024]
  const float* Wo     = (const float*)d_in[5];  // [1024,1024]
  const float* bo     = (const float*)d_in[6];  // [1024]
  float* out = (float*)d_out;                   // [8,4096,1024]

  char* ws = (char*)d_ws;
  const size_t QB = (size_t)8 * 4096 * 1024 * 2;
  const size_t KB = (size_t)8 * 512 * 1024 * 2;
  __bf16* qb = (__bf16*)(ws);
  __bf16* kb = (__bf16*)(ws + QB);
  __bf16* vb = (__bf16*)(ws + QB + KB);
  __bf16* ab = (__bf16*)(ws + QB + 2 * KB);

  const dim3 blk(256);
  gemm128_wmma<false, false><<<dim3(8, 256), blk, 0, stream>>>(
      hidden, Wq, Wq /*unused*/, qb, 32768, 1024, 1024);
  gemm128_wmma<false, false><<<dim3(8, 32), blk, 0, stream>>>(
      enc, Wk, Wk /*unused*/, kb, 4096, 768, 1024);
  gemm128_wmma<false, false><<<dim3(8, 32), blk, 0, stream>>>(
      enc, Wv, Wv /*unused*/, vb, 4096, 768, 1024);
  attn_fused<<<dim3(32, 8, 8), blk, 0, stream>>>(qb, kb, vb, ab);
  gemm128_wmma<true, true><<<dim3(8, 256), blk, 0, stream>>>(
      ab, Wo, bo, out, 32768, 1024, 1024);
}
